// PVConv_88587995447585
// MI455X (gfx1250) — matrix operand order repaired and checked by reading, not compiled, not run on hardware
//
#include <hip/hip_runtime.h>
#include <hip/hip_bf16.h>

#define RR  32
#define R3  32768
#define BB  8
#define CC  64
#define NN  65536
#define NEG_SLOPE 0.1f
#define BN_EPS    1.0e-4f

typedef __attribute__((ext_vector_type(16))) _Float16 v16h;
typedef __attribute__((ext_vector_type(8)))  _Float16 v8h;
typedef __attribute__((ext_vector_type(8)))  float    v8f;
typedef __attribute__((ext_vector_type(4)))  float    v4f;

// ---------------------------------------------------------------------------
// Stage 1a: per-(batch,axis) mean of coords. One block per (b,axis).
// ---------------------------------------------------------------------------
__global__ __launch_bounds__(256) void k_stats_mean(const float* __restrict__ coords,
                                                    float* __restrict__ means) {
    const int b  = blockIdx.x / 3;
    const int ax = blockIdx.x % 3;
    const float* p = coords + ((size_t)b * 3 + ax) * NN;
    float s = 0.0f;
    for (int i = threadIdx.x; i < NN; i += 256) s += p[i];
    __shared__ float red[256];
    red[threadIdx.x] = s;
    __syncthreads();
    for (int off = 128; off > 0; off >>= 1) {
        if (threadIdx.x < off) red[threadIdx.x] += red[threadIdx.x + off];
        __syncthreads();
    }
    if (threadIdx.x == 0) means[blockIdx.x] = red[0] * (1.0f / NN);
}

// ---------------------------------------------------------------------------
// Stage 1b: per-batch max L2 norm of centered coords. One block per batch.
// ---------------------------------------------------------------------------
__global__ __launch_bounds__(256) void k_stats_scale(const float* __restrict__ coords,
                                                     const float* __restrict__ means,
                                                     float* __restrict__ scaleb) {
    const int b = blockIdx.x;
    const float mx = means[b * 3 + 0], my = means[b * 3 + 1], mz = means[b * 3 + 2];
    const float* px = coords + ((size_t)b * 3 + 0) * NN;
    const float* py = coords + ((size_t)b * 3 + 1) * NN;
    const float* pz = coords + ((size_t)b * 3 + 2) * NN;
    float mxsq = 0.0f;
    for (int i = threadIdx.x; i < NN; i += 256) {
        float dx = px[i] - mx, dy = py[i] - my, dz = pz[i] - mz;
        float d2 = dx * dx + dy * dy + dz * dz;
        mxsq = fmaxf(mxsq, d2);
    }
    __shared__ float red[256];
    red[threadIdx.x] = mxsq;
    __syncthreads();
    for (int off = 128; off > 0; off >>= 1) {
        if (threadIdx.x < off) red[threadIdx.x] = fmaxf(red[threadIdx.x], red[threadIdx.x + off]);
        __syncthreads();
    }
    if (threadIdx.x == 0) scaleb[b] = sqrtf(red[0]);
}

// ---------------------------------------------------------------------------
// Stage 2: voxelize scatter. One thread per point; 64 contiguous atomic adds
// into the channel-last sum grid [b][vox][c].
// ---------------------------------------------------------------------------
__global__ __launch_bounds__(256) void k_scatter(const float* __restrict__ features,
                                                 const float* __restrict__ coords,
                                                 const float* __restrict__ means,
                                                 const float* __restrict__ scaleb,
                                                 float* __restrict__ gridsum,
                                                 float* __restrict__ cnt,
                                                 float* __restrict__ vox) {
    const int gid = blockIdx.x * 256 + threadIdx.x;   // BB*NN threads exactly
    const int b = gid >> 16;
    const int n = gid & (NN - 1);
    const float inv2s = 1.0f / (scaleb[b] * 2.0f);
    float s[3];
#pragma unroll
    for (int ax = 0; ax < 3; ++ax) {
        float c = coords[((size_t)b * 3 + ax) * NN + n];
        float nv = (c - means[b * 3 + ax]) * inv2s + 0.5f;
        nv = fminf(fmaxf(nv * (float)RR, 0.0f), (float)(RR - 1));
        s[ax] = nv;
        vox[((size_t)b * 3 + ax) * NN + n] = nv;      // vox_coords for devoxelize
    }
    const int vx = (int)rintf(s[0]);
    const int vy = (int)rintf(s[1]);
    const int vz = (int)rintf(s[2]);
    const int idx = (vx * RR + vy) * RR + vz;
    atomicAdd(&cnt[b * R3 + idx], 1.0f);
    float* gs = gridsum + ((size_t)(b * R3 + idx)) * CC;
    const float* f = features + ((size_t)b * CC) * NN + n;
#pragma unroll 8
    for (int c = 0; c < CC; ++c) atomicAdd(&gs[c], f[(size_t)c * NN]);
}

// ---------------------------------------------------------------------------
// Stage 3: normalize sums by counts and convert to f16 channel-last grid.
// ---------------------------------------------------------------------------
__global__ __launch_bounds__(256) void k_finalize(const float* __restrict__ gridsum,
                                                  const float* __restrict__ cnt,
                                                  _Float16* __restrict__ g0h) {
    const int v = blockIdx.x * 256 + threadIdx.x;     // BB*R3 threads exactly
    const float inv = 1.0f / fmaxf(cnt[v], 1.0f);
    const float* src = gridsum + (size_t)v * CC;
    _Float16* dst = g0h + (size_t)v * CC;
#pragma unroll 8
    for (int c = 0; c < CC; ++c) dst[c] = (_Float16)(src[c] * inv);
}

// ---------------------------------------------------------------------------
// Weight repack: OIDHW f32 -> [tap][co][ci] f16 (B-matrix friendly layout).
// ---------------------------------------------------------------------------
__global__ void k_prep_w(const float* __restrict__ w, _Float16* __restrict__ wh) {
    const int i = blockIdx.x * 256 + threadIdx.x;
    if (i >= 27 * CC * CC) return;
    const int t  = i >> 12;          // tap 0..26
    const int co = (i >> 6) & 63;
    const int ci = i & 63;
    wh[i] = (_Float16)w[((size_t)(co * CC + ci)) * 27 + t];
}

// Fold conv bias + BN into per-channel scale/shift.
__global__ void k_prep_bn(const float* __restrict__ bi, const float* __restrict__ ga,
                          const float* __restrict__ be, const float* __restrict__ mu,
                          const float* __restrict__ va, float* __restrict__ sc,
                          float* __restrict__ sh) {
    const int i = threadIdx.x;
    if (i < CC) {
        float s = ga[i] * rsqrtf(va[i] + BN_EPS);
        sc[i] = s;
        sh[i] = be[i] + (bi[i] - mu[i]) * s;
    }
}

// ---------------------------------------------------------------------------
// Stage 4/5: implicit-GEMM 3x3x3 conv via v_wmma_f32_16x16x32_f16.
// Wave tile: 16 x-positions (M) x 64 out-channels (N), K = 27 taps * 64 in-ch.
// 8 waves/WG -> 4 (b,z,y) rows per workgroup. Channel-last in/out.
//
// Per-tap 64x64 f16 weight block (8 KB) is staged in LDS, double-buffered:
// while tap t computes from ldsw[t&1], tap t+1's block is fetched
// global->regs (one v16h per thread) and stored to ldsw[(t+1)&1] after the
// compute reads. One barrier per tap keeps the two buffers coherent.
// A fragments: four contiguous 16B global loads per lane (one 128B row).
// B fragments: ds_load_b128 pairs from LDS (contiguous k = 16*hi + s, same order as A).
// ---------------------------------------------------------------------------
template <bool OUT16>
__global__ __launch_bounds__(256) void k_conv(const _Float16* __restrict__ gin,
                                              const _Float16* __restrict__ wh,
                                              const float* __restrict__ sc,
                                              const float* __restrict__ sh,
                                              void* __restrict__ gout) {
    __shared__ __align__(32) _Float16 ldsw[2][CC * CC];   // 2 x 8 KB

    const int tid     = threadIdx.x;
    const int wave    = tid >> 5;
    const int lane    = tid & 31;
    const int lane_hi = lane >> 4;
    const int lm      = lane & 15;

    const int row = blockIdx.x * 4 + (wave >> 1);  // 0..BB*RR*RR-1
    const int xh  = wave & 1;
    const int b   = row >> 10;
    const int rem = row & 1023;
    const int z   = rem >> 5;
    const int y   = rem & 31;
    const int x0  = xh * 16;

    // Prologue: stage tap 0 weights (256 threads x 32B = 8 KB).
    v16h wnext = *(const v16h*)(wh + (size_t)tid * 16);
    *(v16h*)&ldsw[0][tid * 16] = wnext;

    v8f acc[4] = {};

    for (int t = 0; t < 27; ++t) {
        __syncthreads();   // ldsw[t&1] ready; all reads of ldsw[(t+1)&1] done

        // Prefetch next tap's weight block into registers (overlaps compute).
        if (t + 1 < 27)
            wnext = *(const v16h*)(wh + (size_t)(t + 1) * CC * CC + (size_t)tid * 16);

        const int dz = t / 9 - 1;
        const int dy = (t / 3) % 3 - 1;
        const int dx = t % 3 - 1;
        const int zz = z + dz;
        const int yy = y + dy;
        if ((unsigned)zz < RR && (unsigned)yy < RR) {   // uniform per wave
            const int  xx    = x0 + lm + dx;
            const bool valid = (unsigned)xx < RR;
            const _Float16* abase =
                gin + (size_t)(((b * RR + zz) * RR + yy) * RR + (valid ? xx : 0)) * CC;

            // All A data for this tap lives in one 128B channel-last row:
            // grouped loads -> one clause, one wait.
            v8h a0 = {}, a1 = {}, a2 = {}, a3 = {};
            if (valid) {
                a0 = *(const v8h*)(abase +      16 * lane_hi);   // cb0: K = 16*hi+0..7
                a1 = *(const v8h*)(abase +  8 + 16 * lane_hi);   // cb0: K = 16*hi+8..15
                a2 = *(const v8h*)(abase + 32 + 16 * lane_hi);   // cb1 lo
                a3 = *(const v8h*)(abase + 40 + 16 * lane_hi);   // cb1 hi
            }

            const _Float16* lw = ldsw[t & 1];
#pragma unroll
            for (int cb = 0; cb < 2; ++cb) {
                v16h a;
#pragma unroll
                for (int i = 0; i < 8; ++i) {
                    a[i]     = cb ? a2[i] : a0[i];
                    a[i + 8] = cb ? a3[i] : a1[i];
                }
                const _Float16* wb = lw + cb * 32 + 16 * lane_hi;
                const v16h b0 = *(const v16h*)(wb + (size_t)(0 * 16 + lm) * CC);
                const v16h b1 = *(const v16h*)(wb + (size_t)(1 * 16 + lm) * CC);
                const v16h b2 = *(const v16h*)(wb + (size_t)(2 * 16 + lm) * CC);
                const v16h b3 = *(const v16h*)(wb + (size_t)(3 * 16 + lm) * CC);

                acc[0] = __builtin_amdgcn_wmma_f32_16x16x32_f16(false, a, false, b0,
                                                                (short)0, acc[0], false, false);
                acc[1] = __builtin_amdgcn_wmma_f32_16x16x32_f16(false, a, false, b1,
                                                                (short)0, acc[1], false, false);
                acc[2] = __builtin_amdgcn_wmma_f32_16x16x32_f16(false, a, false, b2,
                                                                (short)0, acc[2], false, false);
                acc[3] = __builtin_amdgcn_wmma_f32_16x16x32_f16(false, a, false, b3,
                                                                (short)0, acc[3], false, false);
            }
        }

        // Publish next tap's weights to the other LDS buffer.
        if (t + 1 < 27)
            *(v16h*)&ldsw[(t + 1) & 1][tid * 16] = wnext;
    }

    // Epilogue: BN (folded scale/shift) + LeakyReLU, channel-last store.
    const size_t outbase = (size_t)(((b * RR + z) * RR + y) * RR) * CC;
#pragma unroll
    for (int nt = 0; nt < 4; ++nt) {
        const int   co = nt * 16 + lm;
        const float s  = sc[co];
        const float h  = sh[co];
#pragma unroll
        for (int r = 0; r < 8; ++r) {
            const int m = r + 8 * lane_hi;                 // D layout: M = r + 8*laneHi
            float v = acc[nt][r] * s + h;
            v = (v >= 0.0f) ? v : NEG_SLOPE * v;
            const size_t oidx = outbase + (size_t)(x0 + m) * CC + co;
            if (OUT16) ((_Float16*)gout)[oidx] = (_Float16)v;
            else       ((float*)gout)[oidx]    = v;
        }
    }
}

// ---------------------------------------------------------------------------
// Stage 6: trilinear devoxelize + coords passthrough. One thread per point.
// Corner gathers are 64 contiguous channels (float4 vectorized).
// ---------------------------------------------------------------------------
__global__ __launch_bounds__(256) void k_devox(const float* __restrict__ g2,
                                               const float* __restrict__ vox,
                                               const float* __restrict__ coords,
                                               float* __restrict__ out) {
    const int gid = blockIdx.x * 256 + threadIdx.x;   // BB*NN threads exactly
    const int b = gid >> 16;
    const int n = gid & (NN - 1);

    const float sx = vox[((size_t)b * 3 + 0) * NN + n];
    const float sy = vox[((size_t)b * 3 + 1) * NN + n];
    const float sz = vox[((size_t)b * 3 + 2) * NN + n];
    const int x0 = (int)floorf(sx), y0 = (int)floorf(sy), z0 = (int)floorf(sz);
    const float fx = sx - (float)x0, fy = sy - (float)y0, fz = sz - (float)z0;
    const int x1 = min(x0 + 1, RR - 1), y1 = min(y0 + 1, RR - 1), z1 = min(z0 + 1, RR - 1);

    const float w000 = (1 - fx) * (1 - fy) * (1 - fz);
    const float w001 = (1 - fx) * (1 - fy) * fz;
    const float w010 = (1 - fx) * fy * (1 - fz);
    const float w011 = (1 - fx) * fy * fz;
    const float w100 = fx * (1 - fy) * (1 - fz);
    const float w101 = fx * (1 - fy) * fz;
    const float w110 = fx * fy * (1 - fz);
    const float w111 = fx * fy * fz;

    const float* base = g2 + (size_t)b * R3 * CC;
    const float* c000 = base + (size_t)((x0 * RR + y0) * RR + z0) * CC;
    const float* c001 = base + (size_t)((x0 * RR + y0) * RR + z1) * CC;
    const float* c010 = base + (size_t)((x0 * RR + y1) * RR + z0) * CC;
    const float* c011 = base + (size_t)((x0 * RR + y1) * RR + z1) * CC;
    const float* c100 = base + (size_t)((x1 * RR + y0) * RR + z0) * CC;
    const float* c101 = base + (size_t)((x1 * RR + y0) * RR + z1) * CC;
    const float* c110 = base + (size_t)((x1 * RR + y1) * RR + z0) * CC;
    const float* c111 = base + (size_t)((x1 * RR + y1) * RR + z1) * CC;

#pragma unroll 4
    for (int q = 0; q < CC / 4; ++q) {
        v4f a = w000 * (*(const v4f*)(c000 + q * 4)) + w001 * (*(const v4f*)(c001 + q * 4)) +
                w010 * (*(const v4f*)(c010 + q * 4)) + w011 * (*(const v4f*)(c011 + q * 4)) +
                w100 * (*(const v4f*)(c100 + q * 4)) + w101 * (*(const v4f*)(c101 + q * 4)) +
                w110 * (*(const v4f*)(c110 + q * 4)) + w111 * (*(const v4f*)(c111 + q * 4));
#pragma unroll
        for (int j = 0; j < 4; ++j)
            out[((size_t)(b * CC + q * 4 + j)) * NN + n] = a[j];
    }
    // coords passthrough (second tuple element)
    float* out2 = out + (size_t)BB * CC * NN;
#pragma unroll
    for (int ax = 0; ax < 3; ++ax)
        out2[((size_t)b * 3 + ax) * NN + n] = coords[((size_t)b * 3 + ax) * NN + n];
}

// ---------------------------------------------------------------------------
// Host launcher.
// ---------------------------------------------------------------------------
extern "C" void kernel_launch(void* const* d_in, const int* in_sizes, int n_in,
                              void* d_out, int out_size, void* d_ws, size_t ws_size,
                              hipStream_t stream) {
    const float* features = (const float*)d_in[0];
    const float* coords   = (const float*)d_in[1];
    const float* w1  = (const float*)d_in[2];
    const float* b1  = (const float*)d_in[3];
    const float* ga1 = (const float*)d_in[4];
    const float* be1 = (const float*)d_in[5];
    const float* m1  = (const float*)d_in[6];
    const float* v1  = (const float*)d_in[7];
    const float* w2  = (const float*)d_in[8];
    const float* b2  = (const float*)d_in[9];
    const float* ga2 = (const float*)d_in[10];
    const float* be2 = (const float*)d_in[11];
    const float* m2  = (const float*)d_in[12];
    const float* v2  = (const float*)d_in[13];
    float* out = (float*)d_out;

    // Workspace carve-up (all offsets >=256B aligned).
    char* W = (char*)d_ws;
    size_t off = 0;
    float*    gridsum = (float*)(W + off);    off += (size_t)BB * R3 * CC * 4;  // 64 MiB (reused as conv2 out)
    float*    cnt     = (float*)(W + off);    off += (size_t)BB * R3 * 4;       // 1 MiB
    _Float16* g0h     = (_Float16*)(W + off); off += (size_t)BB * R3 * CC * 2;  // 32 MiB
    _Float16* g1h     = (_Float16*)(W + off); off += (size_t)BB * R3 * CC * 2;  // 32 MiB
    float*    vox     = (float*)(W + off);    off += (size_t)BB * 3 * NN * 4;   // 6 MiB
    _Float16* wh1     = (_Float16*)(W + off); off += (size_t)27 * CC * CC * 2;
    _Float16* wh2     = (_Float16*)(W + off); off += (size_t)27 * CC * CC * 2;
    float*    sc1     = (float*)(W + off);    off += 256;
    float*    sh1     = (float*)(W + off);    off += 256;
    float*    sc2     = (float*)(W + off);    off += 256;
    float*    sh2     = (float*)(W + off);    off += 256;
    float*    means   = (float*)(W + off);    off += 256;
    float*    scaleb  = (float*)(W + off);    off += 256;
    float*    g2f     = gridsum;              // conv2 output reuses scatter-sum buffer

    // Zero scatter accumulators each call (deterministic replays).
    hipMemsetAsync(gridsum, 0, (size_t)BB * R3 * CC * 4, stream);
    hipMemsetAsync(cnt,     0, (size_t)BB * R3 * 4, stream);

    // Voxelization statistics.
    k_stats_mean<<<BB * 3, 256, 0, stream>>>(coords, means);
    k_stats_scale<<<BB, 256, 0, stream>>>(coords, means, scaleb);

    // Scatter-average into channel-last grid, normalize -> f16.
    k_scatter<<<(BB * NN) / 256, 256, 0, stream>>>(features, coords, means, scaleb,
                                                   gridsum, cnt, vox);
    k_finalize<<<(BB * R3) / 256, 256, 0, stream>>>(gridsum, cnt, g0h);

    // Weight / BN prep.
    k_prep_w<<<(27 * CC * CC + 255) / 256, 256, 0, stream>>>(w1, wh1);
    k_prep_w<<<(27 * CC * CC + 255) / 256, 256, 0, stream>>>(w2, wh2);
    k_prep_bn<<<1, 64, 0, stream>>>(b1, ga1, be1, m1, v1, sc1, sh1);
    k_prep_bn<<<1, 64, 0, stream>>>(b2, ga2, be2, m2, v2, sc2, sh2);

    // Two WMMA implicit-GEMM convs (conv1 -> f16 grid, conv2 -> f32 grid).
    const int conv_blocks = (BB * RR * RR) / 4;   // 4 rows per workgroup
    k_conv<true><<<conv_blocks, 256, 0, stream>>>(g0h, wh1, sc1, sh1, (void*)g1h);
    k_conv<false><<<conv_blocks, 256, 0, stream>>>(g1h, wh2, sc2, sh2, (void*)g2f);

    // Trilinear devoxelize + coords passthrough.
    k_devox<<<(BB * NN) / 256, 256, 0, stream>>>(g2f, vox, coords, out);
}
